// QuantumMultiHeadAttention_7352984011343
// MI455X (gfx1250) — matrix operand order repaired and checked by reading, not compiled, not run on hardware
//
#include <hip/hip_runtime.h>
#include <hip/hip_bf16.h>
#include <math.h>

typedef _Float16 half_t;
typedef __attribute__((ext_vector_type(16))) _Float16 v16h;
typedef __attribute__((ext_vector_type(8)))  float    v8f;

#define B_  2
#define S_  2048
#define D_  1024
#define H_  16
#define DK_ 64

__device__ __forceinline__ v8f wmma_f16(v16h a, v16h b, v8f c) {
  // 8 args: (neg_a, A, neg_b, B, c_mod, C, reuse_a, reuse_b)
  return __builtin_amdgcn_wmma_f32_16x16x32_f16(false, a, false, b, (short)0, c, false, false);
}

__device__ __forceinline__ v8f zero8() {
  v8f z; for (int i = 0; i < 8; ++i) z[i] = 0.f; return z;
}

// branch-free tanh: 1 - 2/(exp(2x)+1); clamp keeps exp finite
__device__ __forceinline__ float fast_tanh(float x) {
  float xc = fminf(fmaxf(x, -9.f), 9.f);
  float e2 = __expf(2.f * xc);
  float r  = __builtin_amdgcn_rcpf(e2 + 1.f);
  return fmaf(-2.f, r, 1.f);
}

// ---- A-operand fragment (16x32 f16), row-major source, leading dim ld ----
// lane: row m = lane&15, K = {ko..ko+7, 16+ko..16+ko+7}, ko = 8*(lane>>4)
__device__ __forceinline__ v16h load_frag_a(const half_t* tile, int ld, int lane, int kbase) {
  int row = lane & 15;
  int ko  = (lane >> 4) << 3;
  const half_t* p = tile + row * ld + kbase + ko;
  union { v16h v; uint4 q[2]; } u;
  u.q[0] = *(const uint4*)(p);
  u.q[1] = *(const uint4*)(p + 16);
  return u.v;
}

// ---- B-operand fragment (32x16 f16): B[k][n] supplied as T[n][k] row-major ----
// lane: col n = lane&15, K = {ko..ko+7, 16+ko..16+ko+7}, ko = 8*(lane>>4)
__device__ __forceinline__ v16h load_frag_b(const half_t* tileT, int ld, int lane, int kbase) {
  int col = lane & 15;
  int ko  = (lane >> 4) << 3;
  const half_t* p = tileT + col * ld + kbase + ko;
  union { v16h v; uint4 q[2]; } u;
  u.q[0] = *(const uint4*)(p);
  u.q[1] = *(const uint4*)(p + 16);
  return u.v;
}

// ---- A-operand fragment from f32 source (converted to f16) ----
__device__ __forceinline__ v16h load_frag_a_f32(const float* tile, int ld, int lane, int kbase) {
  int row = lane & 15;
  int ko  = (lane >> 4) << 3;
  const float* p = tile + row * ld + kbase + ko;
  float4 x0 = ((const float4*)p)[0];
  float4 x1 = ((const float4*)p)[1];
  float4 y0 = ((const float4*)(p + 16))[0];
  float4 y1 = ((const float4*)(p + 16))[1];
  v16h r;
  r[0]=(half_t)x0.x;  r[1]=(half_t)x0.y;  r[2]=(half_t)x0.z;  r[3]=(half_t)x0.w;
  r[4]=(half_t)x1.x;  r[5]=(half_t)x1.y;  r[6]=(half_t)x1.z;  r[7]=(half_t)x1.w;
  r[8]=(half_t)y0.x;  r[9]=(half_t)y0.y;  r[10]=(half_t)y0.z; r[11]=(half_t)y0.w;
  r[12]=(half_t)y1.x; r[13]=(half_t)y1.y; r[14]=(half_t)y1.z; r[15]=(half_t)y1.w;
  return r;
}

// ============ prep: transpose weights to f16 B-operand form ============
__global__ __launch_bounds__(256) void prep_kernel(
    const float* __restrict__ Wo,
    const float* __restrict__ Wq1, const float* __restrict__ Wq2,
    const float* __restrict__ Wk1, const float* __restrict__ Wk2,
    const float* __restrict__ Wv1, const float* __restrict__ Wv2,
    half_t* __restrict__ Wot, half_t* __restrict__ Wt)
{
  int idx = blockIdx.x * 256 + threadIdx.x;        // 0 .. 1M-1
  int n = idx >> 10, k = idx & 1023;
  Wot[idx] = (half_t)Wo[k * 1024 + n];             // Wot[n][k] = Wo[k][n]
  if (idx < 6 * 4096) {
    int wi = idx >> 12, r = idx & 4095;
    int n2 = r >> 6, k2 = r & 63;
    const float* sp = (wi == 0) ? Wq1 : (wi == 1) ? Wq2 : (wi == 2) ? Wk1
                    : (wi == 3) ? Wk2 : (wi == 4) ? Wv1 : Wv2;
    Wt[idx] = (half_t)sp[k2 * 64 + n2];            // Wt[wi][n][k] = W[k][n]
  }
}

// ============ stage 1: per-head quantum MLP ============
// One wave = one (b,s): 16 head-rows x 64. Out layout [B,H,S,DK] f16.
__global__ __launch_bounds__(256) void quantum_kernel(
    const float* __restrict__ X, const half_t* __restrict__ W1t, const float* __restrict__ b1,
    const half_t* __restrict__ W2t, const float* __restrict__ b2,
    half_t* __restrict__ Out, float outscale)
{
  __shared__ __align__(16) half_t Hl[8][16][72];
  int tid = threadIdx.x, wave = tid >> 5, lane = tid & 31;
  int t = blockIdx.x * 8 + wave;
  int b = t >> 11, s = t & 2047;
  const float* Xrow = X + (size_t)t * D_;

  // preload all W1^T fragments (loop-invariant) so WMMAs don't stall per-tile
  v16h w1f[8];
  for (int nt = 0; nt < 4; ++nt) {
    w1f[2 * nt]     = load_frag_b(W1t + nt * 16 * DK_, DK_, lane, 0);
    w1f[2 * nt + 1] = load_frag_b(W1t + nt * 16 * DK_, DK_, lane, 32);
  }
  v16h ax0 = load_frag_a_f32(Xrow, DK_, lane, 0);
  v16h ax1 = load_frag_a_f32(Xrow, DK_, lane, 32);

  int ncol = lane & 15;
  int mb = (lane < 16) ? 0 : 8;
  half_t (*Hh)[72] = Hl[wave];

  // H1 = tanh(X @ W1 + b1) -> LDS (D-layout -> row-major)
  for (int nt = 0; nt < 4; ++nt) {
    v8f c = zero8();
    c = wmma_f16(ax0, w1f[2 * nt],     c);
    c = wmma_f16(ax1, w1f[2 * nt + 1], c);
    float bias = b1[nt * 16 + ncol];
    for (int j = 0; j < 8; ++j)
      Hh[mb + j][nt * 16 + ncol] = (half_t)fast_tanh(c[j] + bias);
  }
  // preload W2^T fragments while the LDS round-trip drains
  v16h w2f[8];
  for (int nt = 0; nt < 4; ++nt) {
    w2f[2 * nt]     = load_frag_b(W2t + nt * 16 * DK_, DK_, lane, 0);
    w2f[2 * nt + 1] = load_frag_b(W2t + nt * 16 * DK_, DK_, lane, 32);
  }
  // Re-read in A layout (LDS ops same-wave are in-order)
  v16h ah0 = load_frag_a(&Hh[0][0], 72, lane, 0);
  v16h ah1 = load_frag_a(&Hh[0][0], 72, lane, 32);

  float res[4][8];
  for (int nt = 0; nt < 4; ++nt) {
    v8f c = zero8();
    c = wmma_f16(ah0, w2f[2 * nt],     c);
    c = wmma_f16(ah1, w2f[2 * nt + 1], c);
    float bias = b2[nt * 16 + ncol];
    for (int j = 0; j < 8; ++j) res[nt][j] = (c[j] + bias) * outscale;
  }
  for (int nt = 0; nt < 4; ++nt)
    for (int j = 0; j < 8; ++j)
      Hh[mb + j][nt * 16 + ncol] = (half_t)res[nt][j];

  // store rows: 2 lanes per head row; Out[b][h][s][dk]
  int hrow = lane & 15, cseg = lane >> 4;
  half_t* dst = Out + (((size_t)b * H_ + hrow) * S_ + s) * DK_ + cseg * 32;
  const half_t* src = &Hh[hrow][cseg * 32];
  ((uint4*)dst)[0] = ((const uint4*)src)[0];
  ((uint4*)dst)[1] = ((const uint4*)src)[1];
  ((uint4*)dst)[2] = ((const uint4*)src)[2];
  ((uint4*)dst)[3] = ((const uint4*)src)[3];
}

// ============ stage 2: flash attention ============
// grid (S/128, B*H), 8 waves x 16 queries, 64-key tiles, double-buffered LDS.
// Q pre-scaled by 1/sqrt(DK).
__global__ __launch_bounds__(256) void attn_kernel(
    const half_t* __restrict__ Qh, const half_t* __restrict__ Kh,
    const half_t* __restrict__ Vh, half_t* __restrict__ ctx)
{
  __shared__ __align__(16) half_t Kl[2][64][72];    // K tiles, row-major (ping-pong)
  __shared__ __align__(16) half_t Vt[2][64][72];    // V tiles, transposed [dk][key]
  __shared__ __align__(16) half_t Pl[8][16][72];    // per-wave P / out staging

  int tid = threadIdx.x, wave = tid >> 5, lane = tid & 31;
  int bh = blockIdx.y;
  int b = bh >> 4, h = bh & 15;
  int q0 = blockIdx.x * 128;

  const half_t* Qb = Qh + ((size_t)bh * S_ + q0 + wave * 16) * DK_;
  const half_t* Kb = Kh + (size_t)bh * S_ * DK_;
  const half_t* Vb = Vh + (size_t)bh * S_ * DK_;

  v16h aq0 = load_frag_a(Qb, DK_, lane, 0);
  v16h aq1 = load_frag_a(Qb, DK_, lane, 32);

  v8f acc[4];
  for (int d = 0; d < 4; ++d) acc[d] = zero8();
  float mrow[8], lrow[8];
  for (int j = 0; j < 8; ++j) { mrow[j] = -INFINITY; lrow[j] = 0.f; }

  int ncol = lane & 15;
  int mb = (lane < 16) ? 0 : 8;
  half_t (*P)[72] = Pl[wave];
  int ldrow = tid >> 2, ldseg = tid & 3;            // 64 rows x 4 x 16-half segs

  const half_t* kg = Kb + (size_t)ldrow * DK_ + ldseg * 16;
  const half_t* vg = Vb + (size_t)ldrow * DK_ + ldseg * 16;

  // prologue: fetch + stage tile 0 into buffer 0
  uint4 kr0 = ((const uint4*)kg)[0];
  uint4 kr1 = ((const uint4*)kg)[1];
  union { uint4 q[2]; half_t e[16]; } vr;
  vr.q[0] = ((const uint4*)vg)[0];
  vr.q[1] = ((const uint4*)vg)[1];
  *(uint4*)&Kl[0][ldrow][ldseg * 16]     = kr0;
  *(uint4*)&Kl[0][ldrow][ldseg * 16 + 8] = kr1;
  for (int i = 0; i < 16; ++i) Vt[0][ldseg * 16 + i][ldrow] = vr.e[i];
  __syncthreads();

  const int NIT = S_ / 64;
  for (int it = 0; it < NIT; ++it) {
    int cur = it & 1;
    bool more = (it + 1) < NIT;
    // issue next tile's global loads early (overlap with WMMA compute)
    if (more) {
      const half_t* kg2 = kg + (size_t)(it + 1) * 64 * DK_;
      const half_t* vg2 = vg + (size_t)(it + 1) * 64 * DK_;
      kr0 = ((const uint4*)kg2)[0];
      kr1 = ((const uint4*)kg2)[1];
      vr.q[0] = ((const uint4*)vg2)[0];
      vr.q[1] = ((const uint4*)vg2)[1];
    }
    if (it + 2 < NIT) {
      __builtin_prefetch(kg + (size_t)(it + 2) * 64 * DK_, 0, 1);
      __builtin_prefetch(vg + (size_t)(it + 2) * 64 * DK_, 0, 1);
    }

    // scores S = Q @ K^T  (4 n-tiles of 16 keys)
    v8f sc[4];
    for (int nt = 0; nt < 4; ++nt) {
      v8f c = zero8();
      c = wmma_f16(aq0, load_frag_b(&Kl[cur][nt * 16][0], 72, lane, 0),  c);
      c = wmma_f16(aq1, load_frag_b(&Kl[cur][nt * 16][0], 72, lane, 32), c);
      sc[nt] = c;
    }

    // online softmax; rows live across the 16-lane halves of the D layout
    float alpha[8], pv[4][8];
    for (int j = 0; j < 8; ++j) {
      float t = fmaxf(fmaxf(sc[0][j], sc[1][j]), fmaxf(sc[2][j], sc[3][j]));
      t = fmaxf(t, __shfl_xor(t, 1));
      t = fmaxf(t, __shfl_xor(t, 2));
      t = fmaxf(t, __shfl_xor(t, 4));
      t = fmaxf(t, __shfl_xor(t, 8));
      float mn = fmaxf(mrow[j], t);
      float a  = __expf(mrow[j] - mn);
      mrow[j] = mn; alpha[j] = a;
      float rs = 0.f;
      for (int nt = 0; nt < 4; ++nt) {
        float p = __expf(sc[nt][j] - mn);
        pv[nt][j] = p;
        rs += p;
      }
      rs += __shfl_xor(rs, 1);
      rs += __shfl_xor(rs, 2);
      rs += __shfl_xor(rs, 4);
      rs += __shfl_xor(rs, 8);
      lrow[j] = lrow[j] * a + rs;
    }

    // P: D-layout -> row-major LDS, rescale accumulators
    for (int nt = 0; nt < 4; ++nt)
      for (int j = 0; j < 8; ++j)
        P[mb + j][nt * 16 + ncol] = (half_t)pv[nt][j];
    for (int d = 0; d < 4; ++d)
      for (int j = 0; j < 8; ++j) acc[d][j] *= alpha[j];

    // ctx += P @ V  (K=64 keys: 2 WMMAs per dk-tile)
    v16h ap0 = load_frag_a(&P[0][0], 72, lane, 0);
    v16h ap1 = load_frag_a(&P[0][0], 72, lane, 32);
    for (int d = 0; d < 4; ++d) {
      v16h bv0 = load_frag_b(&Vt[cur][d * 16][0], 72, lane, 0);
      v16h bv1 = load_frag_b(&Vt[cur][d * 16][0], 72, lane, 32);
      acc[d] = wmma_f16(ap0, bv0, acc[d]);
      acc[d] = wmma_f16(ap1, bv1, acc[d]);
    }

    // stage next tile into the other buffer (safe: prior readers of that
    // buffer finished before the barrier that ended the previous iteration)
    if (more) {
      int nxt = cur ^ 1;
      *(uint4*)&Kl[nxt][ldrow][ldseg * 16]     = kr0;
      *(uint4*)&Kl[nxt][ldrow][ldseg * 16 + 8] = kr1;
      for (int i = 0; i < 16; ++i) Vt[nxt][ldseg * 16 + i][ldrow] = vr.e[i];
    }
    __syncthreads();
  }

  // epilogue: normalize, transpose via LDS, store ctx [B,S,H,DK] f16
  for (int d = 0; d < 4; ++d)
    for (int j = 0; j < 8; ++j)
      P[mb + j][d * 16 + ncol] = (half_t)(acc[d][j] / lrow[j]);

  int row = lane & 15, cseg = lane >> 4;
  size_t so = (((size_t)b * S_ + q0 + wave * 16 + row) * H_ + h) * DK_ + cseg * 32;
  const half_t* src = &P[row][cseg * 32];
  half_t* dst = ctx + so;
  ((uint4*)dst)[0] = ((const uint4*)src)[0];
  ((uint4*)dst)[1] = ((const uint4*)src)[1];
  ((uint4*)dst)[2] = ((const uint4*)src)[2];
  ((uint4*)dst)[3] = ((const uint4*)src)[3];
}

// ============ stage 3: output projection OUT = CTX @ Wo + bo ============
__global__ __launch_bounds__(256) void outproj_kernel(
    const half_t* __restrict__ A, const half_t* __restrict__ Bt,
    const float* __restrict__ bias, float* __restrict__ Out)
{
  int tid = threadIdx.x, wave = tid >> 5, lane = tid & 31;
  int w  = blockIdx.x * 8 + wave;
  int mt = w & 255, nt4 = w >> 8;                  // 256 M-tiles x 16 (64-wide) N-tiles
  const half_t* Ar = A  + (size_t)mt  * 16 * D_;
  const half_t* Br = Bt + (size_t)nt4 * 64 * D_;

  v8f c[4];
  for (int d = 0; d < 4; ++d) c[d] = zero8();

  // 2-stage software pipeline: fetch k+32 fragments while k's WMMAs run
  v16h a  = load_frag_a(Ar, D_, lane, 0);
  v16h b0 = load_frag_b(Br,             D_, lane, 0);
  v16h b1 = load_frag_b(Br + 16 * D_,   D_, lane, 0);
  v16h b2 = load_frag_b(Br + 32 * D_,   D_, lane, 0);
  v16h b3 = load_frag_b(Br + 48 * D_,   D_, lane, 0);
  for (int k = 32; k < D_; k += 32) {
    v16h an  = load_frag_a(Ar, D_, lane, k);
    v16h bn0 = load_frag_b(Br,           D_, lane, k);
    v16h bn1 = load_frag_b(Br + 16 * D_, D_, lane, k);
    v16h bn2 = load_frag_b(Br + 32 * D_, D_, lane, k);
    v16h bn3 = load_frag_b(Br + 48 * D_, D_, lane, k);
    c[0] = wmma_f16(a, b0, c[0]);
    c[1] = wmma_f16(a, b1, c[1]);
    c[2] = wmma_f16(a, b2, c[2]);
    c[3] = wmma_f16(a, b3, c[3]);
    a = an; b0 = bn0; b1 = bn1; b2 = bn2; b3 = bn3;
  }
  c[0] = wmma_f16(a, b0, c[0]);
  c[1] = wmma_f16(a, b1, c[1]);
  c[2] = wmma_f16(a, b2, c[2]);
  c[3] = wmma_f16(a, b3, c[3]);

  int ncol = lane & 15;
  int mb = (lane < 16) ? 0 : 8;
  for (int d = 0; d < 4; ++d) {
    int col = nt4 * 64 + d * 16 + ncol;
    float bb = bias[col];
    for (int j = 0; j < 8; ++j)
      Out[(size_t)(mt * 16 + mb + j) * D_ + col] = c[d][j] + bb;
  }
}

extern "C" void kernel_launch(void* const* d_in, const int* in_sizes, int n_in,
                              void* d_out, int out_size, void* d_ws, size_t ws_size,
                              hipStream_t stream)
{
  const float* q   = (const float*)d_in[0];
  const float* k   = (const float*)d_in[1];
  const float* v   = (const float*)d_in[2];
  const float* Wq1 = (const float*)d_in[3];
  const float* bq1 = (const float*)d_in[4];
  const float* Wq2 = (const float*)d_in[5];
  const float* bq2 = (const float*)d_in[6];
  const float* Wk1 = (const float*)d_in[7];
  const float* bk1 = (const float*)d_in[8];
  const float* Wk2 = (const float*)d_in[9];
  const float* bk2 = (const float*)d_in[10];
  const float* Wv1 = (const float*)d_in[11];
  const float* bv1 = (const float*)d_in[12];
  const float* Wv2 = (const float*)d_in[13];
  const float* bv2 = (const float*)d_in[14];
  const float* Wo  = (const float*)d_in[15];
  const float* bo  = (const float*)d_in[16];
  float* out = (float*)d_out;

  char* ws = (char*)d_ws;
  half_t* Qh  = (half_t*)(ws);                       // 8 MB  [B,H,S,DK] f16
  half_t* Kh  = (half_t*)(ws + (8u  << 20));         // 8 MB
  half_t* Vh  = (half_t*)(ws + (16u << 20));         // 8 MB
  half_t* CT  = (half_t*)(ws + (24u << 20));         // 8 MB  ctx [B,S,D] f16
  half_t* Wt  = (half_t*)(ws + (32u << 20));         // 48 KB 6x 64x64 f16 (transposed)
  half_t* Wot = (half_t*)(ws + (32u << 20) + (1u << 16)); // 2 MB Wo^T f16

  prep_kernel<<<4096, 256, 0, stream>>>(Wo, Wq1, Wq2, Wk1, Wk2, Wv1, Wv2, Wot, Wt);
  quantum_kernel<<<512, 256, 0, stream>>>(q, Wt,         bq1, Wt + 4096,  bq2, Qh, 0.125f);
  quantum_kernel<<<512, 256, 0, stream>>>(k, Wt + 8192,  bk1, Wt + 12288, bk2, Kh, 1.0f);
  quantum_kernel<<<512, 256, 0, stream>>>(v, Wt + 16384, bv1, Wt + 20480, bv2, Vh, 1.0f);
  attn_kernel<<<dim3(16, 32), 256, 0, stream>>>(Qh, Kh, Vh, CT);
  outproj_kernel<<<512, 256, 0, stream>>>(CT, Wot, bo, out);
}
